// WaveSF_ViT_org_81312320848579
// MI455X (gfx1250) — compile-verified, matrix operand order checked by reading, not run on hardware
//
#include <hip/hip_runtime.h>
#include <hip/hip_bf16.h>
#include <math.h>
#include <stdint.h>

typedef __attribute__((ext_vector_type(16))) __bf16 v16bf;
typedef __attribute__((ext_vector_type(8)))  float  v8f;
typedef __attribute__((ext_vector_type(4)))  unsigned int u32x4;
typedef __attribute__((ext_vector_type(8)))  int i32x8;
typedef __attribute__((ext_vector_type(4)))  int i32x4;

#define DI __device__ __forceinline__

static constexpr int BB      = 16;
static constexpr int T2C     = 8;
static constexpr int SEQ     = 196;   // tokens per (b, t2) slice
static constexpr int SP      = 224;   // SEQ padded to multiple of 32
static constexpr int CC      = 768;
static constexpr int C2      = 1536;
static constexpr int NHEADC  = 12;
static constexpr int HDC     = 64;
static constexpr int MROWS   = BB * T2C * SEQ;   // 25088
static constexpr int TOKENS_IN = 3136;
static constexpr float SCALE_F = 0.125f;         // 1/sqrt(64)

// ---------------------------------------------------------------- WMMA utils
DI v8f wmma_bf16(v16bf a, v16bf b, v8f c) {
  // v_wmma_f32_16x16x32_bf16 : D = A(16x32) * B(32x16) + C(16x16 f32)
  return __builtin_amdgcn_wmma_f32_16x16x32_bf16(false, a, false, b,
                                                 (short)0, c, false, false);
}

union FragU { v16bf v; u32x4 q[2]; __bf16 e[16]; };

// 16-bit A/B fragment: lanes 0-15 hold K={0..7,16..23}, lanes 16-31 K={8..15,24..31}
// With row-major (contiguous-K) storage this is two 16-byte loads.
DI v16bf frag_ld(const __bf16* p, int hi) {
  FragU u;
  const int s = hi ? 8 : 0;
  u.q[0] = *reinterpret_cast<const u32x4*>(p + s);
  u.q[1] = *reinterpret_cast<const u32x4*>(p + s + 16);
  return u.v;
}
DI v16bf frag_zero() {
  FragU u;
  u.q[0] = (u32x4){0u, 0u, 0u, 0u};
  u.q[1] = (u32x4){0u, 0u, 0u, 0u};
  return u.v;
}

// Gather a B fragment from a token-major [tok][64] LDS tile (transposed access).
DI v16bf frag_ld_gather(const __bf16* vlds, int ktbase, int n, int hi) {
  FragU u;
  const int s = hi ? 8 : 0;
#pragma unroll
  for (int e = 0; e < 8; ++e) {
    u.e[e]     = vlds[(ktbase + s + e) * HDC + n];
    u.e[e + 8] = vlds[(ktbase + s + 16 + e) * HDC + n];
  }
  return u.v;
}

DI float gelu_exact(float x) {
  return 0.5f * x * (1.0f + erff(x * 0.70710678118654752f));
}

// ---------------------------------------------------------------- TDM 2D tile load
// Build a D# per CDNA5 ISA ch.8 and issue TENSOR_LOAD_TO_LDS:
//   2D tensor of 2-byte elements, tile (tile_d0 x tile_d1), row stride stride0,
//   destination LDS packed rows of tile_d0 elements.
DI void tdm_load_2d_bf16(unsigned lds_off, const void* gaddr,
                         unsigned tensor_d0, unsigned tensor_d1,
                         unsigned tile_d0, unsigned tile_d1, unsigned stride0) {
  const unsigned long long ga = (unsigned long long)(uintptr_t)gaddr;
  u32x4 g0;
  g0[0] = 1u;                                   // count=1, user descriptor
  g0[1] = lds_off;                              // lds_addr (bytes)
  g0[2] = (unsigned)ga;                         // global_addr[31:0]
  g0[3] = (unsigned)((ga >> 32) & 0x01FFFFFFu)  // global_addr[56:32]
          | 0x80000000u;                        // type=2 ("image")
  i32x8 g1;
  g1[0] = (int)(1u << 16);                      // data_size = 1 -> 2 bytes
  g1[1] = (int)((tensor_d0 & 0xFFFFu) << 16);   // tensor_dim0 low16 at bits 63:48
  g1[2] = (int)(((tensor_d0 >> 16) & 0xFFFFu) | ((tensor_d1 & 0xFFFFu) << 16));
  g1[3] = (int)(((tensor_d1 >> 16) & 0xFFFFu) | ((tile_d0 & 0xFFFFu) << 16));
  g1[4] = (int)(tile_d1 & 0xFFFFu);             // tile_dim1 (tile_dim2 = 0)
  g1[5] = (int)stride0;                         // tensor_dim0_stride low 32
  g1[6] = 0;                                    // stride0 hi / dim1_stride lo
  g1[7] = 0;
  const i32x4 z4 = {0, 0, 0, 0};
#if __clang_major__ >= 23
  const i32x8 z8 = {0, 0, 0, 0, 0, 0, 0, 0};
  __builtin_amdgcn_tensor_load_to_lds(g0, g1, z4, z4, z8, 0);
#else
  __builtin_amdgcn_tensor_load_to_lds(g0, g1, z4, z4, 0);
#endif
}

// ------------------------------------------------------- weight prep (f32 KxN -> bf16 NxK)
__global__ void wconv_k(const float* __restrict__ W, __bf16* __restrict__ Wt, int N) {
  const int K = CC;
  const int idx = blockIdx.x * blockDim.x + threadIdx.x;
  if (idx >= K * N) return;
  const int n = idx / K;
  const int k = idx - n * K;
  Wt[idx] = (__bf16)W[(size_t)k * N + n];
}

// ------------------------------------------------------- permute + convert + row means
__global__ __launch_bounds__(128) void permute_k(const float* __restrict__ xin,
                                                 __bf16* __restrict__ xa,
                                                 __bf16* __restrict__ xd,
                                                 float* __restrict__ ma,
                                                 float* __restrict__ md) {
  const int lane = threadIdx.x & 31;
  const int wave = threadIdx.x >> 5;
  const int m = blockIdx.x * 4 + wave;          // row in [0, MROWS)
  const int s = m % SEQ;
  const int bj = m / SEQ;
  const int j = bj & 7;
  const int b = bj >> 3;
  const int nh = s / 14, nw = s % 14;
#pragma unroll
  for (int par = 0; par < 2; ++par) {
    const int t = 2 * j + par;
    const int token = (((t >> 2) * 14 + nh) * 14 + nw) * 4 + (t & 3);
    const float* src = xin + ((size_t)b * TOKENS_IN + token) * CC;
    __bf16* dst = (par ? xd : xa) + (size_t)m * CC;
    float sum = 0.0f;
    for (int c = lane; c < CC; c += 32) {
      const float v = src[c];
      sum += v;
      dst[c] = (__bf16)v;
    }
    for (int off = 16; off > 0; off >>= 1) sum += __shfl_xor(sum, off, 32);
    if (lane == 0) (par ? md : ma)[m] = sum * (1.0f / 768.0f);
  }
}

// ------------------------------------------------------- bf16 WMMA GEMM, K=768
// A: MROWS x 768 row-major bf16;  Wt: N x 768 row-major bf16 (pre-transposed weights)
// MODE 0: Out = bf16(acc + bias)
// MODE 1: Out = bf16(float(Out) - gelu(acc + bias))     (in-place residual update)
// MODE 2: Out = bf16(gelu(acc + bias))
template <int MODE>
__global__ __launch_bounds__(128) void gemm_bf16_k(const __bf16* __restrict__ A,
                                                   const __bf16* __restrict__ Wt,
                                                   const float* __restrict__ bias,
                                                   __bf16* __restrict__ Out, int N) {
  const int K = CC;
  const int lane = threadIdx.x & 31;
  const int wave = threadIdx.x >> 5;
  const int hi = (lane >= 16) ? 1 : 0;
  const int ln = lane & 15;
  const int rowbase = (blockIdx.x * 4 + wave) * 16;  // MROWS % 64 == 0
  const int colbase = blockIdx.y * 64;

  const __bf16* arow = A + (size_t)(rowbase + ln) * K;
  const __bf16* wrow = Wt + (size_t)(colbase + ln) * K;

  const v8f zer = {0.f, 0.f, 0.f, 0.f, 0.f, 0.f, 0.f, 0.f};
  v8f acc[4] = {zer, zer, zer, zer};

  for (int k = 0; k < K; k += 32) {
    __builtin_prefetch(arow + k + 256, 0, 1);   // global_prefetch_b8
    const v16bf a = frag_ld(arow + k, hi);
#pragma unroll
    for (int t = 0; t < 4; ++t) {
      const v16bf b = frag_ld(wrow + (size_t)(t * 16) * K + k, hi);
      acc[t] = wmma_bf16(a, b, acc[t]);
    }
  }

  const int rl = hi ? 8 : 0;
#pragma unroll
  for (int t = 0; t < 4; ++t) {
    const int col = colbase + t * 16 + ln;
    const float bv = bias[col];
#pragma unroll
    for (int v = 0; v < 8; ++v) {
      const int row = rowbase + rl + v;
      const size_t oidx = (size_t)row * N + col;
      const float x = acc[t][v] + bv;
      if (MODE == 0) {
        Out[oidx] = (__bf16)x;
      } else if (MODE == 1) {
        Out[oidx] = (__bf16)((float)Out[oidx] - gelu_exact(x));
      } else {
        Out[oidx] = (__bf16)gelu_exact(x);
      }
    }
  }
}

// ------------------------------------------------------- attention, one wave per (b,t2,head,qtile)
// V tile is brought into LDS by the Tensor Data Mover (TENSOR_LOAD_TO_LDS),
// overlapped with Q/K fragment loads and the score WMMAs; synchronized with
// s_wait_tensorcnt.
__global__ __launch_bounds__(32) void attn_k(const __bf16* __restrict__ Q,
                                             const __bf16* __restrict__ KV,
                                             __bf16* __restrict__ O) {
  __shared__ __align__(128) __bf16 vtile[SP * HDC];  // token-major [tok][64]
  __shared__ float  sc[16 * SP];                     // score rows for this q-tile
  __shared__ __bf16 pbuf[16 * SP];                   // softmax probs (bf16), zero padded
  __shared__ float  invl[16];

  int idx = blockIdx.x;
  const int qt = idx % 13;   idx /= 13;
  const int head = idx % NHEADC; idx /= NHEADC;   // idx = b*8 + j
  const size_t rowbase = (size_t)idx * SEQ;

  const int lane = threadIdx.x;
  const int hi = (lane >= 16) ? 1 : 0;
  const int ln = lane & 15;
  const int rl = hi ? 8 : 0;

  const __bf16* qbase = Q + rowbase * CC + head * HDC;
  const __bf16* kbase = KV + rowbase * C2 + head * HDC;
  const __bf16* vbase = kbase + CC;   // V half of the KV projection

  // ---- kick off the TDM: V tile (SEQ x 64 bf16, row stride C2) -> LDS, async
  tdm_load_2d_bf16((unsigned)(uintptr_t)(void*)vtile, (const void*)vbase,
                   /*tensor_d0=*/HDC, /*tensor_d1=*/SEQ,
                   /*tile_d0=*/HDC, /*tile_d1=*/SEQ, /*stride0=*/C2);

  // ---- zero the padding token rows (196..223) so 0-probs never meet NaN bits
  for (int i = lane; i < (SP - SEQ) * HDC / 4; i += 32) {
    *reinterpret_cast<unsigned long long*>(vtile + SEQ * HDC + i * 4) = 0ull;
  }

  // ---- Q fragments for this 16-row q tile (K = 64 -> two fragments)
  const int qtok = qt * 16 + ln;
  const __bf16* qrow = qbase + (size_t)qtok * CC;
  const v16bf aq0 = (qtok < SEQ) ? frag_ld(qrow, hi) : frag_zero();
  const v16bf aq1 = (qtok < SEQ) ? frag_ld(qrow + 32, hi) : frag_zero();

  // ---- scores = scale * q @ k^T   (13 column tiles of 16 tokens)
  for (int nt = 0; nt < 13; ++nt) {
    const int ktok = nt * 16 + ln;
    const __bf16* krow = kbase + (size_t)ktok * C2;
    const v16bf bk0 = (ktok < SEQ) ? frag_ld(krow, hi) : frag_zero();
    const v16bf bk1 = (ktok < SEQ) ? frag_ld(krow + 32, hi) : frag_zero();
    v8f s = {0.f, 0.f, 0.f, 0.f, 0.f, 0.f, 0.f, 0.f};
    s = wmma_bf16(aq0, bk0, s);
    s = wmma_bf16(aq1, bk1, s);
#pragma unroll
    for (int v = 0; v < 8; ++v) sc[(rl + v) * SP + nt * 16 + ln] = s[v] * SCALE_F;
  }
  __syncthreads();

  // ---- row softmax (lanes 0..15, one row each); 1/sum folded into the O store
  if (lane < 16) {
    const int r = lane;
    float mx = -1e30f;
    for (int c = 0; c < SEQ; ++c) mx = fmaxf(mx, sc[r * SP + c]);
    float l = 0.0f;
    for (int c = 0; c < SEQ; ++c) {
      const float e = __expf(sc[r * SP + c] - mx);
      l += e;
      pbuf[r * SP + c] = (__bf16)e;
    }
    for (int c = SEQ; c < SP; ++c) pbuf[r * SP + c] = (__bf16)0.0f;
    invl[r] = 1.0f / l;
  }
  // make sure the TDM transfer has landed in LDS before the O-GEMM reads V
  __builtin_amdgcn_s_wait_tensorcnt(0);
  __syncthreads();

  // ---- O = P @ V   (K dim = SP tokens, 7 steps of 32; N = 64 features)
  const v8f zer = {0.f, 0.f, 0.f, 0.f, 0.f, 0.f, 0.f, 0.f};
  v8f o[4] = {zer, zer, zer, zer};
  const __bf16* prow = pbuf + (size_t)ln * SP;
  for (int kt = 0; kt < SP; kt += 32) {
    const v16bf ap = frag_ld(prow + kt, hi);
#pragma unroll
    for (int t = 0; t < 4; ++t) {
      const v16bf bv = frag_ld_gather(vtile, kt, t * 16 + ln, hi);
      o[t] = wmma_bf16(ap, bv, o[t]);
    }
  }

#pragma unroll
  for (int t = 0; t < 4; ++t) {
#pragma unroll
    for (int v = 0; v < 8; ++v) {
      const int r = rl + v;
      const int tq = qt * 16 + r;
      if (tq < SEQ) {
        O[(rowbase + tq) * CC + head * HDC + t * 16 + ln] = (__bf16)(o[t][v] * invl[r]);
      }
    }
  }
}

// ------------------------------------------------------- row mean over bf16 matrix (C=768)
__global__ __launch_bounds__(128) void rowmean_k(const __bf16* __restrict__ X,
                                                 float* __restrict__ out) {
  const int lane = threadIdx.x & 31;
  const int wave = threadIdx.x >> 5;
  const int m = blockIdx.x * 4 + wave;
  const __bf16* row = X + (size_t)m * CC;
  float sum = 0.0f;
  for (int c = lane; c < CC; c += 32) sum += (float)row[c];
  for (int off = 16; off > 0; off >>= 1) sum += __shfl_xor(sum, off, 32);
  if (lane == 0) out[m] = sum * (1.0f / 768.0f);
}

// ------------------------------------------------------- finalize both outputs
__global__ void final_k(const float* __restrict__ ma, const float* __restrict__ md,
                        const float* __restrict__ ra, const float* __restrict__ rdn,
                        float* __restrict__ out) {
  const int i = blockIdx.x * blockDim.x + threadIdx.x;
  if (i < MROWS) {
    // pool_a = 0.5*(ma+md) - (ma + mean(_a));   x_d_out = mean(x_d_new)
    out[i] = 0.5f * (ma[i] + md[i]) - (ma[i] + ra[i]);
    out[MROWS + i] = rdn[i];
  }
}

// ================================================================ launch
extern "C" void kernel_launch(void* const* d_in, const int* in_sizes, int n_in,
                              void* d_out, int out_size, void* d_ws, size_t ws_size,
                              hipStream_t stream) {
  (void)in_sizes; (void)n_in; (void)out_size; (void)ws_size;
  const float* x_in  = (const float*)d_in[0];
  const float* Wq_d  = (const float*)d_in[1];
  const float* bq_d  = (const float*)d_in[2];
  const float* Wkv_a = (const float*)d_in[3];
  const float* bkv_a = (const float*)d_in[4];
  const float* Wq_a  = (const float*)d_in[5];
  const float* bq_a  = (const float*)d_in[6];
  const float* Wkv_d = (const float*)d_in[7];
  const float* bkv_d = (const float*)d_in[8];
  const float* Wp_d  = (const float*)d_in[9];
  const float* bp_d  = (const float*)d_in[10];
  const float* Wp_a  = (const float*)d_in[11];
  const float* bp_a  = (const float*)d_in[12];
  float* out = (float*)d_out;

  char* ws = (char*)d_ws;
  size_t off = 0;
  auto alloc = [&](size_t bytes) -> char* {
    char* p = ws + off;
    off += (bytes + 255) & ~(size_t)255;
    return p;
  };
  __bf16* wt_qd  = (__bf16*)alloc((size_t)CC * CC * 2);
  __bf16* wt_kva = (__bf16*)alloc((size_t)CC * C2 * 2);
  __bf16* wt_qa  = (__bf16*)alloc((size_t)CC * CC * 2);
  __bf16* wt_kvd = (__bf16*)alloc((size_t)CC * C2 * 2);
  __bf16* wt_pd  = (__bf16*)alloc((size_t)CC * CC * 2);
  __bf16* wt_pa  = (__bf16*)alloc((size_t)CC * CC * 2);
  __bf16* xa     = (__bf16*)alloc((size_t)MROWS * CC * 2);
  __bf16* xd     = (__bf16*)alloc((size_t)MROWS * CC * 2);   // becomes x_d_new in place
  __bf16* qbuf   = (__bf16*)alloc((size_t)MROWS * CC * 2);   // q_d, then q_a, then _a
  __bf16* kvbuf  = (__bf16*)alloc((size_t)MROWS * C2 * 2);   // kv_a, then kv_d
  __bf16* attnb  = (__bf16*)alloc((size_t)MROWS * CC * 2);   // attention outputs
  float*  ma     = (float*)alloc((size_t)MROWS * 4);
  float*  md     = (float*)alloc((size_t)MROWS * 4);
  float*  ra     = (float*)alloc((size_t)MROWS * 4);
  float*  rdn    = (float*)alloc((size_t)MROWS * 4);

  // --- weight transpose/convert to bf16 [N][K]
  {
    const int thr = 256;
    wconv_k<<<dim3((CC * CC + thr - 1) / thr), dim3(thr), 0, stream>>>(Wq_d,  wt_qd,  CC);
    wconv_k<<<dim3((CC * C2 + thr - 1) / thr), dim3(thr), 0, stream>>>(Wkv_a, wt_kva, C2);
    wconv_k<<<dim3((CC * CC + thr - 1) / thr), dim3(thr), 0, stream>>>(Wq_a,  wt_qa,  CC);
    wconv_k<<<dim3((CC * C2 + thr - 1) / thr), dim3(thr), 0, stream>>>(Wkv_d, wt_kvd, C2);
    wconv_k<<<dim3((CC * CC + thr - 1) / thr), dim3(thr), 0, stream>>>(Wp_d,  wt_pd,  CC);
    wconv_k<<<dim3((CC * CC + thr - 1) / thr), dim3(thr), 0, stream>>>(Wp_a,  wt_pa,  CC);
  }

  // --- token permutation, bf16 conversion, original row means
  permute_k<<<dim3(MROWS / 4), dim3(128), 0, stream>>>(x_in, xa, xd, ma, md);

  const dim3 gemmBlk(128);
  const dim3 gN1(MROWS / 64, CC / 64);   // 392 x 12
  const dim3 gN2(MROWS / 64, C2 / 64);   // 392 x 24
  const dim3 attnGrid(BB * T2C * NHEADC * 13);

  // --- block 1: d attends to a
  gemm_bf16_k<0><<<gN1, gemmBlk, 0, stream>>>(xd, wt_qd, bq_d, qbuf, CC);
  gemm_bf16_k<0><<<gN2, gemmBlk, 0, stream>>>(xa, wt_kva, bkv_a, kvbuf, C2);
  attn_k<<<attnGrid, dim3(32), 0, stream>>>(qbuf, kvbuf, attnb);
  gemm_bf16_k<1><<<gN1, gemmBlk, 0, stream>>>(attnb, wt_pd, bp_d, xd, CC);  // xd -= gelu(proj)
  rowmean_k<<<dim3(MROWS / 4), dim3(128), 0, stream>>>(xd, rdn);

  // --- block 2: a attends to (updated) d
  gemm_bf16_k<0><<<gN1, gemmBlk, 0, stream>>>(xa, wt_qa, bq_a, qbuf, CC);
  gemm_bf16_k<0><<<gN2, gemmBlk, 0, stream>>>(xd, wt_kvd, bkv_d, kvbuf, C2);
  attn_k<<<attnGrid, dim3(32), 0, stream>>>(qbuf, kvbuf, attnb);
  gemm_bf16_k<2><<<gN1, gemmBlk, 0, stream>>>(attnb, wt_pa, bp_a, qbuf, CC); // _a
  rowmean_k<<<dim3(MROWS / 4), dim3(128), 0, stream>>>(qbuf, ra);

  // --- outputs
  final_k<<<dim3((MROWS + 255) / 256), dim3(256), 0, stream>>>(ma, md, ra, rdn, out);
}
